// InformerBlock_86277303042470
// MI455X (gfx1250) — matrix operand order repaired and checked by reading, not compiled, not run on hardware
//
#include <hip/hip_runtime.h>
#include <hip/hip_bf16.h>

#define BB 4
#define LL 1024
#define DD 768
#define UU 7097
#define DFF 3072
#define MAXC 64

typedef unsigned short u16;
typedef __attribute__((ext_vector_type(16))) __bf16 v16bf;
typedef __attribute__((ext_vector_type(8)))  float  v8f;
typedef __attribute__((ext_vector_type(4)))  int    v4i;

union Frag { uint4 u[2]; v16bf v; };

#if defined(__AMDGCN__) && __has_builtin(__builtin_amdgcn_global_load_async_to_lds_b128) && __has_builtin(__builtin_amdgcn_s_wait_asynccnt)
#define HAS_ASYNC 1
#else
#define HAS_ASYNC 0
#endif

__device__ __forceinline__ void stage_b128(const u16* gp, u16* lp) {
#if HAS_ASYNC
    __builtin_amdgcn_global_load_async_to_lds_b128((v4i*)gp, (v4i*)lp, 0, 0);
#else
    *(uint4*)lp = *(const uint4*)gp;
#endif
}

__device__ __forceinline__ void wait_async() {
#if HAS_ASYNC
    __builtin_amdgcn_s_wait_asynccnt(0);
#endif
}

__device__ __forceinline__ u16 f2bf(float f) {
    unsigned u = __float_as_uint(f);
    unsigned r = (u + 0x7FFFu + ((u >> 16) & 1u)) >> 16;
    return (u16)r;
}

// ---------------- fp32 -> bf16 convert ----------------
__global__ void conv_bf16_kernel(const float* __restrict__ in, u16* __restrict__ out, int n) {
    int i = blockIdx.x * 256 + threadIdx.x;
    if (i < n) out[i] = f2bf(in[i]);
}

// ---------------- BF16 WMMA GEMM: C[M,N] = scale*(A[M,K] @ Bt[N,K]^T) + bias ----------------
// Block tile 128x64, 8 waves as 4(M) x 2(N); each wave computes 32x32 = 4 WMMA tiles.
// Double-buffered LDS, async global->LDS staging (fallback: sync loads), K-step 32.
template <bool GELU>
__global__ __launch_bounds__(256) void gemm_bf16_kernel(
    const u16* __restrict__ A, int lda, long long sA,
    const u16* __restrict__ Bt, int ldb, long long sB,
    float* __restrict__ Cf, u16* __restrict__ Ch, int ldc, long long sC,
    const float* __restrict__ bias, float scale, int K)
{
    __shared__ __align__(16) u16 As[2][128 * 32];
    __shared__ __align__(16) u16 Bs[2][64 * 32];

    const int t  = threadIdx.x;
    const int m0 = blockIdx.y * 128;
    const int n0 = blockIdx.x * 64;
    const int z  = blockIdx.z;
    A  += (size_t)z * sA;
    Bt += (size_t)z * sB;

    // staging map: 256 threads x 16B; A tile 128x32 (2 chunks/thread), B tile 64x32 (1 chunk)
    const int sr = t >> 2;              // 0..63
    const int sc = (t & 3) << 3;        // 0,8,16,24 (bf16 elements)
    const size_t aoff0 = (size_t)(m0 + sr) * lda + sc;
    const size_t aoff1 = (size_t)(m0 + 64 + sr) * lda + sc;
    const size_t boff  = (size_t)(n0 + sr) * ldb + sc;
    const int lA0 = sr * 32 + sc;
    const int lA1 = (64 + sr) * 32 + sc;

    const int wid = t >> 5, lane = t & 31;
    const int wm = wid & 3, wn = wid >> 2;
    const int lm = lane & 15, g = lane >> 4;

    v8f c00 = {}, c01 = {}, c10 = {}, c11 = {};

    const int kIters = K >> 5;
    // prologue: stage K-tile 0 into buffer 0
    stage_b128(&A[aoff0], &As[0][lA0]);
    stage_b128(&A[aoff1], &As[0][lA1]);
    stage_b128(&Bt[boff], &Bs[0][lA0]);

    for (int ki = 0; ki < kIters; ++ki) {
        wait_async();
        __syncthreads();                      // buf[ki&1] visible to all waves
        if (ki + 1 < kIters) {                // software pipeline: stage next tile
            const int k0 = (ki + 1) << 5;
            const int nb = (ki + 1) & 1;
            stage_b128(&A[aoff0 + k0], &As[nb][lA0]);
            stage_b128(&A[aoff1 + k0], &As[nb][lA1]);
            stage_b128(&Bt[boff + k0], &Bs[nb][lA0]);
        }
        const u16* Ab = As[ki & 1];
        const u16* Bb = Bs[ki & 1];

        Frag a0, a1, b0, b1;
        // A frag: halves 0..7 -> K=g*8.., 8..15 -> K=16+g*8..
        const uint4* ap0 = (const uint4*)&Ab[(wm * 32 + lm) * 32];
        a0.u[0] = ap0[g]; a0.u[1] = ap0[2 + g];
        const uint4* ap1 = (const uint4*)&Ab[(wm * 32 + 16 + lm) * 32];
        a1.u[0] = ap1[g]; a1.u[1] = ap1[2 + g];
        // B frag: halves 0..7 -> K=g*8.., 8..15 -> K=16+g*8.. (Bs stored [N][K])
        const uint4* bp0 = (const uint4*)&Bb[(wn * 32 + lm) * 32];
        b0.u[0] = bp0[g]; b0.u[1] = bp0[2 + g];
        const uint4* bp1 = (const uint4*)&Bb[(wn * 32 + 16 + lm) * 32];
        b1.u[0] = bp1[g]; b1.u[1] = bp1[2 + g];

        c00 = __builtin_amdgcn_wmma_f32_16x16x32_bf16(false, a0.v, false, b0.v, (short)0, c00, false, false);
        c01 = __builtin_amdgcn_wmma_f32_16x16x32_bf16(false, a0.v, false, b1.v, (short)0, c01, false, false);
        c10 = __builtin_amdgcn_wmma_f32_16x16x32_bf16(false, a1.v, false, b0.v, (short)0, c10, false, false);
        c11 = __builtin_amdgcn_wmma_f32_16x16x32_bf16(false, a1.v, false, b1.v, (short)0, c11, false, false);
    }

    const int row0 = m0 + wm * 32 + g * 8;    // C layout: VGPR r -> M = r + 8*g
    const int col0 = n0 + wn * 32 + lm;
    const float bb0 = bias ? bias[col0]      : 0.f;
    const float bb1 = bias ? bias[col0 + 16] : 0.f;
    float* Cfp = Cf ? Cf + (size_t)z * sC : nullptr;
    u16*  Chp = Ch ? Ch + (size_t)z * sC : nullptr;

    auto emit = [&](float acc, int row, int col, float bb) {
        float v0 = acc * scale + bb;
        if (GELU) v0 = 0.5f * v0 * (1.f + erff(v0 * 0.70710678118f));
        size_t o = (size_t)row * ldc + col;
        if (Cfp) Cfp[o] = v0;
        if (Chp) Chp[o] = f2bf(v0);
    };
#pragma unroll
    for (int r = 0; r < 8; ++r) {
        emit(c00[r], row0 + r,      col0,      bb0);
        emit(c01[r], row0 + r,      col0 + 16, bb1);
        emit(c10[r], row0 + 16 + r, col0,      bb0);
        emit(c11[r], row0 + 16 + r, col0 + 16, bb1);
    }
}

// ---------------- per-query-row selected-set bitmap: count + complement list (batch independent) ----
__global__ __launch_bounds__(256) void maskstat_kernel(const int* __restrict__ idx,
                                                       int* __restrict__ mcnt, int* __restrict__ mlist) {
    const int i = blockIdx.x, t = threadIdx.x;
    __shared__ unsigned bm[LL / 32];
    __shared__ int cnt;
    if (t < LL / 32) bm[t] = 0u;
    if (t == 0) cnt = 0;
    __syncthreads();
    for (int u = t; u < UU; u += 256) {
        int p = idx[(size_t)i * UU + u];
        atomicOr(&bm[p >> 5], 1u << (p & 31));
    }
    __syncthreads();
    for (int p = t; p < LL; p += 256) {
        if (!((bm[p >> 5] >> (p & 31)) & 1u)) {
            int pos = atomicAdd(&cnt, 1);
            if (pos < MAXC) mlist[i * MAXC + pos] = p;
        }
    }
    __syncthreads();
    if (t == 0) mcnt[i] = cnt;
}

// ---------------- row gather: max/mean over idx -> softmax coefficients ----------------
__global__ __launch_bounds__(256) void rowstat_kernel(const float* __restrict__ full,
                                                      const int* __restrict__ idx,
                                                      const int* __restrict__ mcnt,
                                                      float* __restrict__ coefA, float* __restrict__ coefB) {
    const int bi = blockIdx.x;           // b*L + i
    const int i  = bi & (LL - 1);
    const int t  = threadIdx.x;
    __shared__ float row[LL];
    __shared__ float rmax[256], rsum[256];
    ((float4*)row)[t] = ((const float4*)(full + (size_t)bi * LL))[t];
    __syncthreads();
    float mx = -3.4e38f, sm = 0.f;
    for (int u = t; u < UU; u += 256) {
        float vv = row[idx[(size_t)i * UU + u]];
        mx = fmaxf(mx, vv);
        sm += vv;
    }
    rmax[t] = mx; rsum[t] = sm;
    __syncthreads();
    for (int s = 128; s > 0; s >>= 1) {
        if (t < s) { rmax[t] = fmaxf(rmax[t], rmax[t + s]); rsum[t] += rsum[t + s]; }
        __syncthreads();
    }
    if (t == 0) {
        float mm = rmax[0] - rsum[0] * (1.f / (float)UU);
        float e  = expf(mm);
        int   m  = mcnt[i];
        float c  = (float)(LL - m);
        float Z  = c * e + (float)m;      // c*exp(mm) + (L-c)*exp(0)
        coefA[bi] = e / Z;                // weight for Vsum
        coefB[bi] = (1.f - e) / Z;        // weight for Vunsel
    }
}

// ---------------- column sums of v per batch ----------------
__global__ __launch_bounds__(256) void vsum_kernel(const float* __restrict__ v, float* __restrict__ vs) {
    const int b = blockIdx.x / 3, seg = blockIdx.x % 3;
    const int d = seg * 256 + threadIdx.x;
    float acc = 0.f;
    const float* p = v + (size_t)b * LL * DD + d;
    for (int l = 0; l < LL; ++l) acc += p[(size_t)l * DD];
    vs[b * DD + d] = acc;
}

// ---------------- z + residual + LayerNorm1 (fp32 + bf16 outputs) ----------------
__global__ __launch_bounds__(256) void zln1_kernel(const float* __restrict__ x, const float* __restrict__ v,
                                                   const float* __restrict__ vs,
                                                   const float* __restrict__ coefA, const float* __restrict__ coefB,
                                                   const int* __restrict__ mcnt, const int* __restrict__ mlist,
                                                   const float* __restrict__ g1, const float* __restrict__ be1,
                                                   float* __restrict__ x1, u16* __restrict__ x1h) {
    const int bi = blockIdx.x, t = threadIdx.x;
    const int b = bi >> 10, i = bi & 1023;
    __shared__ float red[256];
    __shared__ float stat[2];
    const float a  = coefA[bi];
    const float bc = coefB[bi];
    int m = mcnt[i]; if (m > MAXC) m = MAXC;
    float tv[3];
#pragma unroll
    for (int j = 0; j < 3; ++j) {
        int d = t + j * 256;
        float acc = 0.f;
        for (int e2 = 0; e2 < m; ++e2) {
            int p = mlist[i * MAXC + e2];
            acc += v[((size_t)b * LL + p) * DD + d];
        }
        tv[j] = x[(size_t)bi * DD + d] + a * vs[b * DD + d] + bc * acc;
    }
    red[t] = tv[0] + tv[1] + tv[2];
    __syncthreads();
    for (int s = 128; s > 0; s >>= 1) { if (t < s) red[t] += red[t + s]; __syncthreads(); }
    if (t == 0) stat[0] = red[0] * (1.f / (float)DD);
    __syncthreads();
    const float mu = stat[0];
    float vv = 0.f;
#pragma unroll
    for (int j = 0; j < 3; ++j) { float d0 = tv[j] - mu; vv += d0 * d0; }
    red[t] = vv;
    __syncthreads();
    for (int s = 128; s > 0; s >>= 1) { if (t < s) red[t] += red[t + s]; __syncthreads(); }
    if (t == 0) stat[1] = rsqrtf(red[0] * (1.f / (float)DD) + 1e-5f);
    __syncthreads();
    const float rs = stat[1];
#pragma unroll
    for (int j = 0; j < 3; ++j) {
        int d = t + j * 256;
        float o = (tv[j] - mu) * rs * g1[d] + be1[d];
        x1[(size_t)bi * DD + d]  = o;
        x1h[(size_t)bi * DD + d] = f2bf(o);
    }
}

// ---------------- residual + LayerNorm2 -> out ----------------
__global__ __launch_bounds__(256) void ln2_kernel(const float* __restrict__ x1, const float* __restrict__ ff,
                                                  const float* __restrict__ g2, const float* __restrict__ be2,
                                                  float* __restrict__ out) {
    const int bi = blockIdx.x, t = threadIdx.x;
    __shared__ float red[256];
    __shared__ float stat[2];
    float tv[3];
#pragma unroll
    for (int j = 0; j < 3; ++j) {
        int d = t + j * 256;
        tv[j] = x1[(size_t)bi * DD + d] + ff[(size_t)bi * DD + d];
    }
    red[t] = tv[0] + tv[1] + tv[2];
    __syncthreads();
    for (int s = 128; s > 0; s >>= 1) { if (t < s) red[t] += red[t + s]; __syncthreads(); }
    if (t == 0) stat[0] = red[0] * (1.f / (float)DD);
    __syncthreads();
    const float mu = stat[0];
    float vv = 0.f;
#pragma unroll
    for (int j = 0; j < 3; ++j) { float d0 = tv[j] - mu; vv += d0 * d0; }
    red[t] = vv;
    __syncthreads();
    for (int s = 128; s > 0; s >>= 1) { if (t < s) red[t] += red[t + s]; __syncthreads(); }
    if (t == 0) stat[1] = rsqrtf(red[0] * (1.f / (float)DD) + 1e-5f);
    __syncthreads();
    const float rs = stat[1];
#pragma unroll
    for (int j = 0; j < 3; ++j) {
        int d = t + j * 256;
        out[(size_t)bi * DD + d] = (tv[j] - mu) * rs * g2[d] + be2[d];
    }
}

extern "C" void kernel_launch(void* const* d_in, const int* in_sizes, int n_in,
                              void* d_out, int out_size, void* d_ws, size_t ws_size,
                              hipStream_t stream) {
    const float* x   = (const float*)d_in[0];
    const int*   idx = (const int*)d_in[1];
    const float* Wq  = (const float*)d_in[2];
    const float* bq  = (const float*)d_in[3];
    const float* Wk  = (const float*)d_in[4];
    const float* bk  = (const float*)d_in[5];
    const float* Wv  = (const float*)d_in[6];
    const float* bv  = (const float*)d_in[7];
    const float* W1  = (const float*)d_in[8];
    const float* b1  = (const float*)d_in[9];
    const float* W2  = (const float*)d_in[10];
    const float* b2  = (const float*)d_in[11];
    const float* g1  = (const float*)d_in[12];
    const float* be1 = (const float*)d_in[13];
    const float* g2  = (const float*)d_in[14];
    const float* be2 = (const float*)d_in[15];
    float* out = (float*)d_out;

    const size_t ML = (size_t)BB * LL;      // 4096 rows
    char* ws = (char*)d_ws;
    size_t off = 0;
    auto alloc = [&](size_t bytes) -> void* {
        void* p = ws + off;
        off = (off + bytes + 255) & ~(size_t)255;
        return p;
    };

    u16*   xh   = (u16*)alloc(ML * DD * 2);
    u16*   Wqh  = (u16*)alloc((size_t)DD * DD * 2);
    u16*   Wkh  = (u16*)alloc((size_t)DD * DD * 2);
    u16*   Wvh  = (u16*)alloc((size_t)DD * DD * 2);
    u16*   W1h  = (u16*)alloc((size_t)DFF * DD * 2);
    u16*   W2h  = (u16*)alloc((size_t)DD * DFF * 2);
    float* vbuf = (float*)alloc(ML * DD * 4);
    float* vs   = (float*)alloc((size_t)BB * DD * 4);
    int*   mcnt = (int*)alloc((size_t)LL * 4);
    int*   mlist= (int*)alloc((size_t)LL * MAXC * 4);
    float* cfA  = (float*)alloc(ML * 4);
    float* cfB  = (float*)alloc(ML * 4);
    float* x1   = (float*)alloc(ML * DD * 4);
    u16*   x1h  = (u16*)alloc(ML * DD * 2);
    float* ff2  = (float*)alloc(ML * DD * 4);
    // overlapped region: {qh, kh, full} reused later as ffh (lifetimes disjoint)
    size_t regionQK = ML * DD * 2 * 2 + (size_t)BB * LL * LL * 4;   // qh + kh + full
    size_t regionFF = ML * DFF * 2;                                 // ffh
    char*  region   = (char*)alloc(regionQK > regionFF ? regionQK : regionFF);
    u16*   qh   = (u16*)region;
    u16*   kh   = qh + ML * DD;
    float* full = (float*)(region + ML * DD * 2 * 2);
    u16*   ffh  = (u16*)region;

    dim3 blk(256);
    auto cv = [&](const float* src, u16* dst, size_t n) {
        conv_bf16_kernel<<<dim3((unsigned)((n + 255) / 256)), blk, 0, stream>>>(src, dst, (int)n);
    };
    cv(x,  xh,  ML * DD);
    cv(Wq, Wqh, (size_t)DD * DD);
    cv(Wk, Wkh, (size_t)DD * DD);
    cv(Wv, Wvh, (size_t)DD * DD);
    cv(W1, W1h, (size_t)DFF * DD);
    cv(W2, W2h, (size_t)DD * DFF);

    // QKV projections (weights stored [N,K] row-major == transposed layout)
    gemm_bf16_kernel<false><<<dim3(DD / 64, ML / 128, 1), blk, 0, stream>>>(
        xh, DD, 0, Wqh, DD, 0, nullptr, qh, DD, 0, bq, 1.f, DD);
    gemm_bf16_kernel<false><<<dim3(DD / 64, ML / 128, 1), blk, 0, stream>>>(
        xh, DD, 0, Wkh, DD, 0, nullptr, kh, DD, 0, bk, 1.f, DD);
    gemm_bf16_kernel<false><<<dim3(DD / 64, ML / 128, 1), blk, 0, stream>>>(
        xh, DD, 0, Wvh, DD, 0, vbuf, nullptr, DD, 0, bv, 1.f, DD);

    // full = scale * q @ k^T, batched over B
    const float scale = 1.f / sqrtf((float)DD);
    gemm_bf16_kernel<false><<<dim3(LL / 64, LL / 128, BB), blk, 0, stream>>>(
        qh, DD, (long long)LL * DD, kh, DD, (long long)LL * DD,
        full, nullptr, LL, (long long)LL * LL, nullptr, scale, DD);

    // selected-set stats (batch independent) and per-(b,i) softmax coefficients
    maskstat_kernel<<<dim3(LL), blk, 0, stream>>>(idx, mcnt, mlist);
    rowstat_kernel<<<dim3((unsigned)ML), blk, 0, stream>>>(full, idx, mcnt, cfA, cfB);
    vsum_kernel<<<dim3(BB * 3), blk, 0, stream>>>(vbuf, vs);

    // z = attn@v via two-value softmax decomposition, + residual + LN1
    zln1_kernel<<<dim3((unsigned)ML), blk, 0, stream>>>(
        x, vbuf, vs, cfA, cfB, mcnt, mlist, g1, be1, x1, x1h);

    // FFN
    gemm_bf16_kernel<true><<<dim3(DFF / 64, ML / 128, 1), blk, 0, stream>>>(
        x1h, DD, 0, W1h, DD, 0, nullptr, ffh, DFF, 0, b1, 1.f, DD);
    gemm_bf16_kernel<false><<<dim3(DD / 64, ML / 128, 1), blk, 0, stream>>>(
        ffh, DFF, 0, W2h, DFF, 0, ff2, nullptr, DD, 0, b2, 1.f, DFF);

    // residual + LN2 -> out
    ln2_kernel<<<dim3((unsigned)ML), blk, 0, stream>>>(x1, ff2, g2, be2, out);
}
